// Model_67405216743703
// MI455X (gfx1250) — matrix operand order repaired and checked by reading, not compiled, not run on hardware
//
#include <hip/hip_runtime.h>
#include <math.h>

// ============================================================================
// Informer-style encoder/decoder for MI455X (gfx1250).
// Dense math: WMMA f16->f32 GEMM (v_wmma_f32_16x16x32_f16), operands are
// pre-converted to f16 (B pre-transposed to NxK) so GEMM tiles are staged
// into LDS with the Tensor Data Mover (tensor_load_to_lds, TENSORcnt), with
// LDS row padding generated by the D# pad fields (32 dwords + 4 dwords pad
// -> 72-half pitch). Falls back to plain b128 copies if the TDM builtin is
// unavailable. Model is compute-bound (~100+ GFLOP vs ~100 MB @ 23.3 TB/s),
// so everything funnels into the matrix pipe.
// ============================================================================

typedef __attribute__((ext_vector_type(16))) _Float16 v16h;
typedef __attribute__((ext_vector_type(8)))  _Float16 v8h;
typedef __attribute__((ext_vector_type(4)))  _Float16 v4h;
typedef __attribute__((ext_vector_type(8)))  float    v8f;
typedef __attribute__((ext_vector_type(4)))  unsigned int u32x4;
typedef __attribute__((ext_vector_type(8)))  int      i32x8;
typedef __attribute__((ext_vector_type(4)))  int      i32x4;

#define BM 128
#define BN 128
#define BK 64
#define KP 72    // LDS row pitch in halfs: 64 data + 8 pad = 144B (16B multiple)
#define NT 256   // threads per block (8 waves)

#if __has_builtin(__builtin_amdgcn_tensor_load_to_lds)
#define USE_TDM 1
#else
#define USE_TDM 0
#endif

__device__ __forceinline__ int imin(int a, int b) { return a < b ? a : b; }

// ---------------------------------------------------------------------------
// WMMA fragment loads from LDS (f16, KP pitch).
// A (16x32, MxK): lane m=lane&15 holds row m; kbase=(lane>>4)*8;
//   elements 0..7 -> K=kbase..kbase+7, elements 8..15 -> K=kbase+16..kbase+23.
// B (32x16, KxN) stored as Bs[n][k]: lane n=lane&15, kbase=(lane>>4)*8;
//   elements 0..7 -> K=kbase..kbase+7, elements 8..15 -> K=kbase+16..kbase+23.
// ---------------------------------------------------------------------------
__device__ __forceinline__ v16h frag_a(const _Float16* tile, int row0, int kk) {
  int lane = threadIdx.x & 31;
  const _Float16* r = tile + (row0 + (lane & 15)) * KP + kk + ((lane >> 4) << 3);
  v8h lo = *(const v8h*)r;
  v8h hi = *(const v8h*)(r + 16);
  v16h f;
#pragma unroll
  for (int i = 0; i < 8; ++i) { f[i] = lo[i]; f[8 + i] = hi[i]; }
  return f;
}

__device__ __forceinline__ v16h frag_b(const _Float16* tile, int col0, int kk) {
  int lane = threadIdx.x & 31;
  const _Float16* r = tile + (col0 + (lane & 15)) * KP + kk + ((lane >> 4) << 3);
  v8h lo = *(const v8h*)r;
  v8h hi = *(const v8h*)(r + 16);
  v16h f;
#pragma unroll
  for (int i = 0; i < 8; ++i) { f[i] = lo[i]; f[8 + i] = hi[i]; }
  return f;
}

// ---------------------------------------------------------------------------
// TDM: DMA one BKx(rows) f16 tile (row-major, row_stride elems) into LDS at
// lds_byte_off with a 72-half padded pitch. 2D descriptor (groups 2/3 zero).
// data_size=2B (code 1); pad_enable; pad_interval code 4 (=32 dwords);
// pad_amount code 3 (=4 dwords). tile_dim0=BK elems, tile_dim1=rows.
// OOB rows/cols zero-fill via tensor_dim = remaining extent.
// ---------------------------------------------------------------------------
__device__ __forceinline__ void tdm_tile_to_lds(unsigned lds_byte_off,
                                                const _Float16* g,
                                                long long row_stride,
                                                int rows_rem, int cols_rem) {
#if USE_TDM
  unsigned long long ga = (unsigned long long)(uintptr_t)g;
  u32x4 g0;
  g0[0] = 1u;                                   // count=1 (valid user D#)
  g0[1] = lds_byte_off;                         // lds_addr
  g0[2] = (unsigned)(ga & 0xffffffffu);         // global_addr[31:0]
  g0[3] = (unsigned)((ga >> 32) & 0x01ffffffu) | (2u << 30);  // [56:32] | type=2
  unsigned td0 = (unsigned)cols_rem;            // tensor_dim0 (elems)
  unsigned td1 = (unsigned)rows_rem;            // tensor_dim1
  unsigned long long st0 = (unsigned long long)row_stride;  // dim0 stride (elems)
  i32x8 g1;
  g1[0] = (int)((1u << 16) | (1u << 20) | (4u << 22) | (3u << 25));
  g1[1] = (int)((td0 & 0xffffu) << 16);                       // dim0[15:0] @63:48
  g1[2] = (int)(((td0 >> 16) & 0xffffu) | ((td1 & 0xffffu) << 16));
  g1[3] = (int)(((td1 >> 16) & 0xffffu) | ((unsigned)BK << 16));  // tile_dim0
  g1[4] = (int)128u;                                          // tile_dim1=128
  g1[5] = (int)(unsigned)(st0 & 0xffffffffu);                 // stride0 lo32
  g1[6] = (int)(unsigned)((st0 >> 32) & 0xffffu);             // stride0 hi16
  g1[7] = 0;
  i32x4 z4 = {};
#if defined(__clang_major__) && (__clang_major__ >= 23)
  i32x8 z8 = {};
  __builtin_amdgcn_tensor_load_to_lds(g0, g1, z4, z4, z8, 0);
#else
  __builtin_amdgcn_tensor_load_to_lds(g0, g1, z4, z4, 0);
#endif
#else
  (void)lds_byte_off; (void)g; (void)row_stride; (void)rows_rem; (void)cols_rem;
#endif
}

// ---------------------------------------------------------------------------
// Batched f16 GEMM: C[bh] = act(alpha * A[bh] x B[bh]^T-layout + bias).
// A: MxK row-major f16 (lda). B: NxK row-major f16 (ldb) — i.e. already
// transposed, so both operands stream contiguously along K.
// z -> (b = z/H, h = z%H); per-tensor (sXb, sXh) element strides allow
// per-head (L,64) views of packed (B*L,512) tensors.
// 256 threads; 128x128x64 tiles; wave (r,c)=(wave>>1,wave&1) owns 32x64 =
// 2x4 WMMA accumulators; 16 v_wmma per K-block.
// ---------------------------------------------------------------------------
template <int ACT>
__global__ __launch_bounds__(NT) void gemm16(
    const _Float16* __restrict__ A, const _Float16* __restrict__ Bm,
    float* __restrict__ C, const float* __restrict__ bias,
    int M, int N, int K, int lda, int ldb, int ldc,
    long long sAb, long long sAh, long long sBb, long long sBh,
    long long sCb, long long sCh, int H, float alpha) {
  __shared__ _Float16 smem[(BM + BN) * KP];
  _Float16* As = smem;
  _Float16* Bs = smem + BM * KP;

  int bz = blockIdx.z;
  int bb = bz / H, hh = bz % H;
  A  += (long long)bb * sAb + (long long)hh * sAh;
  Bm += (long long)bb * sBb + (long long)hh * sBh;
  C  += (long long)bb * sCb + (long long)hh * sCh;

  int n0 = blockIdx.x * BN, m0 = blockIdx.y * BM;
  int tid = threadIdx.x;
  int wave = tid >> 5;
  int wm = (wave >> 1) * 32;   // 4 wave-rows
  int wn = (wave & 1) * 64;    // 2 wave-cols

  v8f zero = {};
  v8f acc[2][4];
#pragma unroll
  for (int i = 0; i < 2; ++i)
#pragma unroll
    for (int j = 0; j < 4; ++j) acc[i][j] = zero;

  for (int k0 = 0; k0 < K; k0 += BK) {
#if USE_TDM
    // Wave 0 DMAs both tiles via the Tensor Data Mover; everyone barriers.
    if (tid < 32) {
      tdm_tile_to_lds(0u, A + (long long)m0 * lda + k0, lda, M - m0, K - k0);
      tdm_tile_to_lds((unsigned)(BM * KP * 2),
                      Bm + (long long)n0 * ldb + k0, ldb, N - n0, K - k0);
      __builtin_amdgcn_s_wait_tensorcnt(0);
    }
    __syncthreads();
#else
    // Fallback: branch-free two-phase b128 copies (loads clustered).
    v8h ra[4], rb[4];
#pragma unroll
    for (int i = 0; i < 4; ++i) {
      int slot = tid + i * NT;               // 1024 slots = 128 rows x 8 v8h
      int r = slot >> 3, c8 = (slot & 7) << 3;
      int gm = imin(m0 + r, M - 1);
      ra[i] = *(const v8h*)(A + (long long)gm * lda + (k0 + c8));
    }
#pragma unroll
    for (int i = 0; i < 4; ++i) {
      int slot = tid + i * NT;
      int n = slot >> 3, c8 = (slot & 7) << 3;
      int gn = imin(n0 + n, N - 1);
      rb[i] = *(const v8h*)(Bm + (long long)gn * ldb + (k0 + c8));
    }
#pragma unroll
    for (int i = 0; i < 4; ++i) {
      int slot = tid + i * NT;
      int r = slot >> 3, c8 = (slot & 7) << 3;
      *(v8h*)(As + r * KP + c8) = ra[i];
    }
#pragma unroll
    for (int i = 0; i < 4; ++i) {
      int slot = tid + i * NT;
      int n = slot >> 3, c8 = (slot & 7) << 3;
      *(v8h*)(Bs + n * KP + c8) = rb[i];
    }
    __syncthreads();
#endif

    // Prefetch next K-block of A (gfx1250 global_prefetch path).
    if (k0 + BK < K) {
      int ra2 = imin(m0 + (tid >> 1), M - 1);
      int ca2 = k0 + BK + ((tid & 1) << 5);
      __builtin_prefetch(&A[(long long)ra2 * lda + ca2], 0, 1);
    }

    // ---- 2 k-steps x (2x4) WMMA tiles. ----
#pragma unroll
    for (int kk = 0; kk < BK; kk += 32) {
      v16h a0 = frag_a(As, wm, kk);
      v16h a1 = frag_a(As, wm + 16, kk);
      v16h b0 = frag_b(Bs, wn, kk);
      v16h b1 = frag_b(Bs, wn + 16, kk);
      v16h b2 = frag_b(Bs, wn + 32, kk);
      v16h b3 = frag_b(Bs, wn + 48, kk);
      acc[0][0] = __builtin_amdgcn_wmma_f32_16x16x32_f16(false, a0, false, b0, (short)0, acc[0][0], false, false);
      acc[0][1] = __builtin_amdgcn_wmma_f32_16x16x32_f16(false, a0, false, b1, (short)0, acc[0][1], false, false);
      acc[0][2] = __builtin_amdgcn_wmma_f32_16x16x32_f16(false, a0, false, b2, (short)0, acc[0][2], false, false);
      acc[0][3] = __builtin_amdgcn_wmma_f32_16x16x32_f16(false, a0, false, b3, (short)0, acc[0][3], false, false);
      acc[1][0] = __builtin_amdgcn_wmma_f32_16x16x32_f16(false, a1, false, b0, (short)0, acc[1][0], false, false);
      acc[1][1] = __builtin_amdgcn_wmma_f32_16x16x32_f16(false, a1, false, b1, (short)0, acc[1][1], false, false);
      acc[1][2] = __builtin_amdgcn_wmma_f32_16x16x32_f16(false, a1, false, b2, (short)0, acc[1][2], false, false);
      acc[1][3] = __builtin_amdgcn_wmma_f32_16x16x32_f16(false, a1, false, b3, (short)0, acc[1][3], false, false);
    }
    __syncthreads();
  }

  // ---- Epilogue. C/D layout: VGPR v -> row v (lanes 0-15) / v+8 (16-31). ----
  int lane = tid & 31;
  int cn = lane & 15, rb8 = (lane >> 4) * 8;
#pragma unroll
  for (int i = 0; i < 2; ++i)
#pragma unroll
    for (int j = 0; j < 4; ++j)
#pragma unroll
      for (int v = 0; v < 8; ++v) {
        int gm = m0 + wm + i * 16 + rb8 + v;
        int gn = n0 + wn + j * 16 + cn;
        if (gm < M && gn < N) {
          float x = acc[i][j][v] * alpha;
          if (bias) x += bias[gn];
          if (ACT == 1)  // exact GELU
            x = 0.5f * x * (1.0f + erff(x * 0.7071067811865476f));
          C[(long long)gm * ldc + gn] = x;
        }
      }
}

// ---------------------------------------------------------------------------
// f32 -> f16 flat convert (n multiple of 4).
// ---------------------------------------------------------------------------
__global__ void cvt_kernel(const float* __restrict__ in, _Float16* __restrict__ out,
                           long long n4) {
  long long i = (long long)blockIdx.x * blockDim.x + threadIdx.x;
  if (i >= n4) return;
  float4 v = ((const float4*)in)[i];
  v4h h;
  h[0] = (_Float16)v.x; h[1] = (_Float16)v.y;
  h[2] = (_Float16)v.z; h[3] = (_Float16)v.w;
  ((v4h*)out)[i] = h;
}

// Transpose+convert: in is a KxN f32 view (ldin, batch strides); out is packed
// per-z NxK f16: out[z][n][k] = in[b,h][k][n].
__global__ void cvtT_kernel(const float* __restrict__ in, _Float16* __restrict__ out,
                            int N, int K, int ldin,
                            long long sInb, long long sInh, int H) {
  int bz = blockIdx.z;
  int b = bz / H, h = bz % H;
  const float* src = in + (long long)b * sInb + (long long)h * sInh;
  _Float16* dst = out + (long long)bz * N * K;
  long long idx = (long long)blockIdx.x * blockDim.x + threadIdx.x;
  long long tot = (long long)N * K;
  if (idx >= tot) return;
  int k = (int)(idx % K);
  int n = (int)(idx / K);
  dst[idx] = (_Float16)src[(long long)k * ldin + n];
}

// ---------------------------------------------------------------------------
// Embedding: circular conv1d (kernel 3, C_IN=7) + sinusoid PE + mark @ Wtemp.
// ---------------------------------------------------------------------------
__global__ void embed_kernel(const float* __restrict__ x,
                             const float* __restrict__ mark,
                             const float* __restrict__ Wtok,
                             const float* __restrict__ Wtemp,
                             float* __restrict__ out,
                             int Bn, int L, int Cin, int Mk) {
  long long idx = (long long)blockIdx.x * blockDim.x + threadIdx.x;
  long long total = (long long)Bn * L * 512;
  if (idx >= total) return;
  int d = (int)(idx % 512);
  int l = (int)((idx / 512) % L);
  int b = (int)(idx / ((long long)512 * L));
  float acc = 0.0f;
  for (int t = 0; t < 3; ++t) {
    int ls = l + t - 1; ls = (ls + L) % L;
    const float* xr = x + ((long long)b * L + ls) * Cin;
    const float* wr = Wtok + (long long)t * Cin * 512 + d;
    for (int c = 0; c < Cin; ++c) acc += xr[c] * wr[(long long)c * 512];
  }
  const float* mr = mark + ((long long)b * L + l) * Mk;
  for (int m = 0; m < Mk; ++m) acc += mr[m] * Wtemp[(long long)m * 512 + d];
  int i2 = d & ~1;
  float f = __expf(-(float)i2 * (logf(10000.0f) / 512.0f));
  float ang = (float)l * f;
  acc += (d & 1) ? cosf(ang) : sinf(ang);
  out[idx] = acc;
}

// ---------------------------------------------------------------------------
// Block-wide reductions (128 threads).
// ---------------------------------------------------------------------------
__device__ __forceinline__ float block_sum128(float v, float* red) {
  int t = threadIdx.x;
  red[t] = v; __syncthreads();
  for (int o = 64; o > 0; o >>= 1) {
    if (t < o) red[t] += red[t + o];
    __syncthreads();
  }
  float r = red[0]; __syncthreads();
  return r;
}
__device__ __forceinline__ float block_max128(float v, float* red) {
  int t = threadIdx.x;
  red[t] = v; __syncthreads();
  for (int o = 64; o > 0; o >>= 1) {
    if (t < o) red[t] = fmaxf(red[t], red[t + o]);
    __syncthreads();
  }
  float r = red[0]; __syncthreads();
  return r;
}

// Row softmax with optional causal mask (query index = row % LQ).
__global__ __launch_bounds__(128) void softmax_kernel(float* __restrict__ S,
                                                      int cols, int LQ, int causal) {
  __shared__ float red[128];
  long long row = blockIdx.x;
  int q = (int)(row % (long long)LQ);
  float* r = S + row * (long long)cols;
  float m = -3.0e38f;
  for (int c = threadIdx.x; c < cols; c += 128) {
    float v = r[c];
    if (causal && c > q) v = -3.0e38f;
    m = fmaxf(m, v);
  }
  m = block_max128(m, red);
  float sum = 0.0f;
  for (int c = threadIdx.x; c < cols; c += 128) {
    float v = (causal && c > q) ? 0.0f : __expf(r[c] - m);
    r[c] = v;
    sum += v;
  }
  sum = block_sum128(sum, red);
  float inv = 1.0f / sum;
  for (int c = threadIdx.x; c < cols; c += 128) r[c] *= inv;
}

// X = LayerNorm(X + Y) * g + b   (Y may be null; d=512, one block per row).
__global__ __launch_bounds__(128) void add_ln_kernel(float* __restrict__ X,
                                                     const float* __restrict__ Y,
                                                     const float* __restrict__ g,
                                                     const float* __restrict__ bta) {
  __shared__ float red[128];
  long long row = blockIdx.x;
  float* xr = X + row * 512;
  const float* yr = Y ? Y + row * 512 : nullptr;
  float s[4];
  float sum = 0.0f;
#pragma unroll
  for (int i = 0; i < 4; ++i) {
    int c = threadIdx.x + i * 128;
    float v = xr[c] + (yr ? yr[c] : 0.0f);
    s[i] = v; sum += v;
  }
  float mean = block_sum128(sum, red) * (1.0f / 512.0f);
  float sq = 0.0f;
#pragma unroll
  for (int i = 0; i < 4; ++i) { float d = s[i] - mean; sq += d * d; }
  float var = block_sum128(sq, red) * (1.0f / 512.0f);
  float rstd = rsqrtf(var + 1e-5f);
#pragma unroll
  for (int i = 0; i < 4; ++i) {
    int c = threadIdx.x + i * 128;
    xr[c] = (s[i] - mean) * rstd * g[c] + bta[c];
  }
}

// im2col for kernel-3 circular conv over d=512: XC[b,l, t*512+c] = X[b,(l+t-1)%L, c]
__global__ void im2col3_kernel(const float* __restrict__ X, float* __restrict__ XC, int Bn, int L) {
  long long idx = (long long)blockIdx.x * blockDim.x + threadIdx.x;
  long long total = (long long)Bn * L * 1536;
  if (idx >= total) return;
  int c = (int)(idx % 1536);
  int l = (int)((idx / 1536) % L);
  int b = (int)(idx / ((long long)1536 * L));
  int t = c / 512, cc = c % 512;
  int ls = (l + t - 1 + L) % L;
  XC[idx] = X[((long long)b * L + ls) * 512 + cc];
}

// BN-scale + ELU + maxpool(window 3, stride 2, pad ±1) : out length = L/2.
__global__ void distill_pool_kernel(const float* __restrict__ Y,
                                    const float* __restrict__ g,
                                    const float* __restrict__ bb,
                                    float* __restrict__ out, int Bn, int L) {
  int Lo = L / 2;
  long long idx = (long long)blockIdx.x * blockDim.x + threadIdx.x;
  long long total = (long long)Bn * Lo * 512;
  if (idx >= total) return;
  int d = (int)(idx % 512);
  int i = (int)((idx / 512) % Lo);
  int b = (int)(idx / ((long long)512 * Lo));
  float m = -1e30f;
#pragma unroll
  for (int t = -1; t <= 1; ++t) {
    int j = 2 * i + t;
    if (j >= 0 && j < L) {
      float v = Y[((long long)b * L + j) * 512 + d] * g[d] + bb[d];
      v = v > 0.0f ? v : (__expf(v) - 1.0f);   // ELU
      m = fmaxf(m, v);
    }
  }
  out[idx] = m;
}

// Final projection 512 -> 1.
__global__ void final_proj_kernel(const float* __restrict__ X,
                                  const float* __restrict__ W,
                                  const float* __restrict__ bp,
                                  float* __restrict__ out, int rows) {
  int r = blockIdx.x * blockDim.x + threadIdx.x;
  if (r >= rows) return;
  const float* xr = X + (long long)r * 512;
  float s = bp[0];
  for (int d = 0; d < 512; ++d) s += xr[d] * W[d];
  out[r] = s;
}

// ===========================================================================
// Host orchestration
// ===========================================================================
struct AttnW { const float *Wq, *bq, *Wk, *bk, *Wv, *bv, *Wo, *bo; };

static inline void cvt(hipStream_t st, _Float16* dst, const float* src, long long n) {
  long long n4 = n >> 2;
  cvt_kernel<<<dim3((unsigned)((n4 + 255) / 256)), dim3(256), 0, st>>>(src, dst, n4);
}
static inline void cvtT(hipStream_t st, _Float16* dst, const float* src,
                        int N, int K, int ldin, long long sb, long long sh,
                        int nb, int H) {
  long long tot = (long long)N * K;
  dim3 g((unsigned)((tot + 255) / 256), 1, nb * H);
  cvtT_kernel<<<g, dim3(256), 0, st>>>(src, dst, N, K, ldin, sb, sh, H);
}
static inline void gemm(hipStream_t st, const _Float16* A, const _Float16* B,
                        float* C, const float* bias, int M, int N, int K,
                        int lda, int ldb, int ldc,
                        long long sAb, long long sAh, long long sBb, long long sBh,
                        long long sCb, long long sCh, int nb, int H,
                        float alpha, int act) {
  dim3 grid((N + BN - 1) / BN, (M + BM - 1) / BM, nb * H);
  dim3 blk(NT);
  if (act == 1)
    gemm16<1><<<grid, blk, 0, st>>>(A, B, C, bias, M, N, K, lda, ldb, ldc,
                                    sAb, sAh, sBb, sBh, sCb, sCh, H, alpha);
  else
    gemm16<0><<<grid, blk, 0, st>>>(A, B, C, bias, M, N, K, lda, ldb, ldc,
                                    sAb, sAh, sBb, sBh, sCb, sCh, H, alpha);
}

struct Bufs {
  float *Q, *Kb, *V, *AO, *T, *H1, *XC, *S;
  _Float16 *Xh, *Xkvh, *Qh, *Kh, *Sh, *AOh, *H1h, *XCh, *Vt, *Wt;
};

// Dense multi-head attention (8 heads, dh=64) -> T = proj output.
static void run_attention(hipStream_t st, const float* X, const float* XKV,
                          int Bn, int L, int Lk, const AttnW& w, int causal, Bufs& u) {
  int BL = Bn * L, BLk = Bn * Lk;
  cvt(st, u.Xh, X, (long long)BL * 512);
  const _Float16* Xkvh = u.Xh;
  if (XKV != X) { cvt(st, u.Xkvh, XKV, (long long)BLk * 512); Xkvh = u.Xkvh; }

  cvtT(st, u.Wt, w.Wq, 512, 512, 512, 0, 0, 1, 1);
  gemm(st, u.Xh, u.Wt, u.Q, w.bq, BL, 512, 512, 512, 512, 512, 0,0,0,0,0,0, 1,1, 1.0f, 0);
  cvtT(st, u.Wt, w.Wk, 512, 512, 512, 0, 0, 1, 1);
  gemm(st, Xkvh, u.Wt, u.Kb, w.bk, BLk, 512, 512, 512, 512, 512, 0,0,0,0,0,0, 1,1, 1.0f, 0);
  cvtT(st, u.Wt, w.Wv, 512, 512, 512, 0, 0, 1, 1);
  gemm(st, Xkvh, u.Wt, u.V, w.bv, BLk, 512, 512, 512, 512, 512, 0,0,0,0,0,0, 1,1, 1.0f, 0);

  // S[b,h] = Q[b,h] x K[b,h]^T / sqrt(64). K (per head Lk x 64) is already the
  // NxK form under a strided f16 view of the packed buffer.
  cvt(st, u.Qh, u.Q, (long long)BL * 512);
  cvt(st, u.Kh, u.Kb, (long long)BLk * 512);
  gemm(st, u.Qh, u.Kh, u.S, nullptr, L, Lk, 64, 512, 512, Lk,
       (long long)L * 512, 64, (long long)Lk * 512, 64,
       (long long)8 * L * Lk, (long long)L * Lk, Bn, 8, 0.125f, 0);
  softmax_kernel<<<Bn * 8 * L, 128, 0, st>>>(u.S, Lk, L, causal);

  // AO[b,:,h*64:] = softmax(S) x V; V transposed per-head to NxK (64 x Lk).
  cvt(st, u.Sh, u.S, (long long)Bn * 8 * L * Lk);
  cvtT(st, u.Vt, u.V, 64, Lk, 512, (long long)Lk * 512, 64, Bn, 8);
  gemm(st, u.Sh, u.Vt, u.AO, nullptr, L, 64, Lk, Lk, Lk, 512,
       (long long)8 * L * Lk, (long long)L * Lk,
       (long long)8 * 64 * Lk, (long long)64 * Lk,
       (long long)L * 512, 64, Bn, 8, 1.0f, 0);

  // T = AO @ Wo + bo
  cvt(st, u.AOh, u.AO, (long long)BL * 512);
  cvtT(st, u.Wt, w.Wo, 512, 512, 512, 0, 0, 1, 1);
  gemm(st, u.AOh, u.Wt, u.T, w.bo, BL, 512, 512, 512, 512, 512, 0,0,0,0,0,0, 1,1, 1.0f, 0);
}

extern "C" void kernel_launch(void* const* d_in, const int* in_sizes, int n_in,
                              void* d_out, int out_size, void* d_ws, size_t ws_size,
                              hipStream_t stream) {
  (void)in_sizes; (void)n_in; (void)out_size; (void)ws_size;
  const int B = 8, LE = 1024, LD = 512;

  int p = 0;
  const float* x_enc      = (const float*)d_in[p++];
  const float* x_mark_enc = (const float*)d_in[p++];
  const float* x_dec      = (const float*)d_in[p++];
  const float* x_mark_dec = (const float*)d_in[p++];
  p++;  // cn (unused int scalar)

  // Params assumed flattened in dict insertion order (see reference setup_inputs).
  auto nf = [&]() { return (const float*)d_in[p++]; };
  const float* encWtok  = nf(); const float* encWtemp = nf();
  const float* decWtok  = nf(); const float* decWtemp = nf();

  AttnW encA[3];
  const float *eW1[3], *eb1[3], *eW2[3], *eb2[3], *el1g[3], *el1b[3], *el2g[3], *el2b[3];
  for (int i = 0; i < 3; ++i) {
    encA[i] = { nf(), nf(), nf(), nf(), nf(), nf(), nf(), nf() };
    eW1[i] = nf(); eb1[i] = nf(); eW2[i] = nf(); eb2[i] = nf();
    el1g[i] = nf(); el1b[i] = nf(); el2g[i] = nf(); el2b[i] = nf();
  }
  const float *cWc[2], *cbc[2], *cg[2], *cb[2];
  for (int i = 0; i < 2; ++i) { cWc[i] = nf(); cbc[i] = nf(); cg[i] = nf(); cb[i] = nf(); }
  const float* encNg = nf(); const float* encNb = nf();

  AttnW decS[2], decC[2];
  const float *dW1[2], *db1[2], *dW2[2], *db2[2];
  const float *dl1g[2], *dl1b[2], *dl2g[2], *dl2b[2], *dl3g[2], *dl3b[2];
  for (int i = 0; i < 2; ++i) {
    decS[i] = { nf(), nf(), nf(), nf(), nf(), nf(), nf(), nf() };
    decC[i] = { nf(), nf(), nf(), nf(), nf(), nf(), nf(), nf() };
    dW1[i] = nf(); db1[i] = nf(); dW2[i] = nf(); db2[i] = nf();
    dl1g[i] = nf(); dl1b[i] = nf(); dl2g[i] = nf(); dl2b[i] = nf();
    dl3g[i] = nf(); dl3b[i] = nf();
  }
  const float* decNg = nf(); const float* decNb = nf();
  const float* Wproj = nf(); const float* bproj = nf();

  // Workspace carve-out.
  char* base = (char*)d_ws;
  size_t off = 0;
  auto allocf = [&](size_t n) { float* q = (float*)(base + off); off += n * 4; return q; };
  auto alloch = [&](size_t n) {
    _Float16* q = (_Float16*)(base + off);
    off += n * 2; off = (off + 15) & ~(size_t)15;
    return q;
  };
  float* encX = allocf((size_t)B * LE * 512);
  float* decX = allocf((size_t)B * LD * 512);
  Bufs u;
  u.Q  = allocf((size_t)B * LE * 512);
  u.Kb = allocf((size_t)B * LE * 512);
  u.V  = allocf((size_t)B * LE * 512);
  u.AO = allocf((size_t)B * LE * 512);
  u.T  = allocf((size_t)B * LE * 512);
  u.H1 = allocf((size_t)B * LE * 2048);
  u.XC = allocf((size_t)B * LE * 1536);
  u.S  = allocf((size_t)B * 8 * LE * LE);
  u.Xh   = alloch((size_t)B * LE * 512);
  u.Xkvh = alloch((size_t)B * LE * 512);
  u.Qh   = alloch((size_t)B * LE * 512);
  u.Kh   = alloch((size_t)B * LE * 512);
  u.AOh  = alloch((size_t)B * LE * 512);
  u.H1h  = alloch((size_t)B * LE * 2048);
  u.XCh  = alloch((size_t)B * LE * 1536);
  u.Vt   = alloch((size_t)B * LE * 512);
  u.Wt   = alloch((size_t)2048 * 512);
  u.Sh   = alloch((size_t)B * 8 * LE * LE);

  dim3 b256(256);
  auto blocks = [](long long n) { return dim3((unsigned)((n + 255) / 256)); };

  // ---------------- Encoder ----------------
  embed_kernel<<<blocks((long long)B * LE * 512), b256, 0, stream>>>(
      x_enc, x_mark_enc, encWtok, encWtemp, encX, B, LE, 7, 4);

  int L = LE;
  for (int i = 0; i < 3; ++i) {
    run_attention(stream, encX, encX, B, L, L, encA[i], /*causal=*/0, u);
    add_ln_kernel<<<B * L, 128, 0, stream>>>(encX, u.T, el1g[i], el1b[i]);
    // FFN
    cvt(stream, u.Xh, encX, (long long)B * L * 512);
    cvtT(stream, u.Wt, eW1[i], 2048, 512, 2048, 0, 0, 1, 1);
    gemm(stream, u.Xh, u.Wt, u.H1, eb1[i], B * L, 2048, 512, 512, 512, 2048,
         0,0,0,0,0,0, 1,1, 1.0f, /*gelu*/1);
    cvt(stream, u.H1h, u.H1, (long long)B * L * 2048);
    cvtT(stream, u.Wt, eW2[i], 512, 2048, 512, 0, 0, 1, 1);
    gemm(stream, u.H1h, u.Wt, u.T, eb2[i], B * L, 512, 2048, 2048, 2048, 512,
         0,0,0,0,0,0, 1,1, 1.0f, 0);
    add_ln_kernel<<<B * L, 128, 0, stream>>>(encX, u.T, el2g[i], el2b[i]);
    if (i < 2) {  // conv distill: im2col GEMM (K=1536) + BN/ELU/maxpool
      im2col3_kernel<<<blocks((long long)B * L * 1536), b256, 0, stream>>>(encX, u.XC, B, L);
      cvt(stream, u.XCh, u.XC, (long long)B * L * 1536);
      cvtT(stream, u.Wt, cWc[i], 512, 1536, 512, 0, 0, 1, 1);
      gemm(stream, u.XCh, u.Wt, u.T, cbc[i], B * L, 512, 1536, 1536, 1536, 512,
           0,0,0,0,0,0, 1,1, 1.0f, 0);
      distill_pool_kernel<<<blocks((long long)B * (L / 2) * 512), b256, 0, stream>>>(
          u.T, cg[i], cb[i], encX, B, L);
      L /= 2;
    }
  }
  add_ln_kernel<<<B * L, 128, 0, stream>>>(encX, nullptr, encNg, encNb);
  int Lenc = L;  // 256

  // ---------------- Decoder ----------------
  embed_kernel<<<blocks((long long)B * LD * 512), b256, 0, stream>>>(
      x_dec, x_mark_dec, decWtok, decWtemp, decX, B, LD, 7, 4);
  for (int i = 0; i < 2; ++i) {
    run_attention(stream, decX, decX, B, LD, LD, decS[i], /*causal=*/1, u);
    add_ln_kernel<<<B * LD, 128, 0, stream>>>(decX, u.T, dl1g[i], dl1b[i]);
    run_attention(stream, decX, encX, B, LD, Lenc, decC[i], /*causal=*/0, u);
    add_ln_kernel<<<B * LD, 128, 0, stream>>>(decX, u.T, dl2g[i], dl2b[i]);
    // FFN
    cvt(stream, u.Xh, decX, (long long)B * LD * 512);
    cvtT(stream, u.Wt, dW1[i], 2048, 512, 2048, 0, 0, 1, 1);
    gemm(stream, u.Xh, u.Wt, u.H1, db1[i], B * LD, 2048, 512, 512, 512, 2048,
         0,0,0,0,0,0, 1,1, 1.0f, 1);
    cvt(stream, u.H1h, u.H1, (long long)B * LD * 2048);
    cvtT(stream, u.Wt, dW2[i], 512, 2048, 512, 0, 0, 1, 1);
    gemm(stream, u.H1h, u.Wt, u.T, db2[i], B * LD, 512, 2048, 2048, 2048, 512,
         0,0,0,0,0,0, 1,1, 1.0f, 0);
    add_ln_kernel<<<B * LD, 128, 0, stream>>>(decX, u.T, dl3g[i], dl3b[i]);
  }
  add_ln_kernel<<<B * LD, 128, 0, stream>>>(decX, nullptr, decNg, decNb);

  final_proj_kernel<<<blocks(B * LD), b256, 0, stream>>>(decX, Wproj, bproj,
                                                         (float*)d_out, B * LD);
}
